// Attention_60516089200696
// MI455X (gfx1250) — compile-verified, matrix operand order checked by reading, not run on hardware
//
#include <hip/hip_runtime.h>

// ---------------- problem constants ----------------
#define DK    1024      // model / head dim
#define S_LEN 4096      // sequence length
#define B_SZ  4         // batch
#define M_TOT (B_SZ * S_LEN)   // 16384 flattened rows

typedef __bf16 bf16x16 __attribute__((ext_vector_type(16)));
typedef float  f32x8   __attribute__((ext_vector_type(8)));
typedef unsigned short ushort8 __attribute__((ext_vector_type(8)));
typedef int    v4i     __attribute__((vector_size(16)));   // matches builtin proto

// Async global->LDS path (gfx1250). Guarded so the file compiles either way.
#if defined(__has_builtin)
#  if __has_builtin(__builtin_amdgcn_global_load_async_to_lds_b128)
#    define USE_ASYNC_LDS 1
#  endif
#endif
#ifndef USE_ASYNC_LDS
#  define USE_ASYNC_LDS 0
#endif

static __device__ __forceinline__ void stage_b128(const unsigned short* gsrc,
                                                  unsigned short* ldst) {
#if USE_ASYNC_LDS
    typedef __attribute__((address_space(1))) v4i* gvp;
    typedef __attribute__((address_space(3))) v4i* lvp;
    __builtin_amdgcn_global_load_async_to_lds_b128((gvp)gsrc, (lvp)ldst, 0, 0);
#else
    *(ushort8*)ldst = *(const ushort8*)gsrc;
#endif
}

static __device__ __forceinline__ void wait_async_lds() {
#if USE_ASYNC_LDS
    asm volatile("s_wait_asynccnt 0x0" ::: "memory");
#endif
}

// float -> bf16, round-to-nearest-even
static __device__ __forceinline__ unsigned short f2bf(float f) {
    unsigned int x = __float_as_uint(f);
    unsigned int r = (x + 0x7FFFu + ((x >> 16) & 1u)) >> 16;
    return (unsigned short)r;
}

// Load one 16x32 bf16 A-fragment (or 32x16 B-fragment from a row-major
// "columns are memory rows" matrix) into the ISA VGPR layout:
//   lane L: memory row = L%16
//   half-index 0..7  -> K = (L<16 ? 0 : 8)  + h
//   half-index 8..15 -> K = (L<16 ? 16: 24) + (h-8)
static __device__ __forceinline__ bf16x16 load_frag(const unsigned short* base, int ld) {
    const int lane = threadIdx.x & 31;
    const int r    = lane & 15;
    const int k0   = (lane & 16) ? 8 : 0;
    const unsigned short* p = base + (size_t)r * ld + k0;
    union { ushort8 h[2]; bf16x16 v; } u;
    u.h[0] = *(const ushort8*)(p);
    u.h[1] = *(const ushort8*)(p + 16);
    return u.v;
}

static __device__ __forceinline__ f32x8 wmma_bf16(bf16x16 a, bf16x16 b, f32x8 c) {
    // (neg_a, A, neg_b, B, c_mod, C, reuse_a, reuse_b)
    return __builtin_amdgcn_wmma_f32_16x16x32_bf16(false, a, false, b, (short)0, c, false, false);
}

// ---------------- fp32 -> bf16 cast ----------------
__global__ void cast_f32_bf16(const float* __restrict__ in, unsigned short* __restrict__ out,
                              long long n) {
    long long i = (long long)blockIdx.x * blockDim.x + threadIdx.x;
    long long stride = (long long)gridDim.x * blockDim.x;
    for (; i < n; i += stride) out[i] = f2bf(in[i]);
}

// ---------------- 128x64-per-block GEMM, LDS-staged B, async double-buffer ----
// C[m][n] = sum_k A[m][k] * W[n][k] + bias[n]
// 8 waves; wave w owns rows m0+w*16 .. +15.  B k-slab (64 n x 32 k) shared in LDS.
// mode 0: bf16 row-major out (ld=N); mode 1: bf16 transposed (Vt[b][n][s]); mode 2: f32.
__global__ __launch_bounds__(256) void gemm128x64(const unsigned short* __restrict__ A,
                                                  const unsigned short* __restrict__ W,
                                                  const float* __restrict__ bias,
                                                  void* __restrict__ outp,
                                                  int N, int K, int mode) {
    __shared__ unsigned short ldsB[2][64 * 32];   // 2 x 4 KB

    const int w    = threadIdx.x >> 5;
    const int m0   = blockIdx.x * 128 + w * 16;
    const int n0   = blockIdx.y * 64;
    const int lane = threadIdx.x & 31;
    const int nn   = lane & 15;
    const int mo   = (lane & 16) ? 8 : 0;

    // staging assignment: thread t copies 16B: B-row srow, element offset sseg
    const int srow = threadIdx.x >> 2;        // 0..63
    const int sseg = (threadIdx.x & 3) * 8;   // 0,8,16,24 (elements)
    const unsigned short* gB = W + (size_t)(n0 + srow) * K + sseg;
    unsigned short* lB[2] = { &ldsB[0][srow * 32 + sseg], &ldsB[1][srow * 32 + sseg] };

    stage_b128(gB, lB[0]);                    // prologue: k-slab 0 -> buf 0

    f32x8 c0 = {}, c1 = {}, c2 = {}, c3 = {};
    const unsigned short* Arow = A + (size_t)m0 * K;
    const int nk = K / 32;
    for (int i = 0; i < nk; ++i) {
        const int kc = i * 32;
        wait_async_lds();                     // my slab-i bytes have landed
        __syncthreads();                      // everyone's landed; iter i-1 compute done
        if (i + 1 < nk) stage_b128(gB + kc + 32, lB[(i + 1) & 1]);
        const unsigned short* curB = &ldsB[i & 1][0];
        bf16x16 a = load_frag(Arow + kc, K);
        c0 = wmma_bf16(a, load_frag(curB +  0 * 32, 32), c0);
        c1 = wmma_bf16(a, load_frag(curB + 16 * 32, 32), c1);
        c2 = wmma_bf16(a, load_frag(curB + 32 * 32, 32), c2);
        c3 = wmma_bf16(a, load_frag(curB + 48 * 32, 32), c3);
    }

    f32x8 cc[4] = {c0, c1, c2, c3};
    for (int t = 0; t < 4; ++t) {
        const int ncol = n0 + t * 16 + nn;
        const float bv = bias ? bias[ncol] : 0.0f;
        for (int i = 0; i < 8; ++i) {
            const float v = cc[t][i] + bv;
            const int mrow = m0 + mo + i;
            if (mode == 0) {
                ((unsigned short*)outp)[(size_t)mrow * N + ncol] = f2bf(v);
            } else if (mode == 1) {
                const int bb = mrow / S_LEN, ss = mrow % S_LEN;
                ((unsigned short*)outp)[((size_t)bb * DK + ncol) * S_LEN + ss] = f2bf(v);
            } else {
                ((float*)outp)[(size_t)mrow * N + ncol] = v;
            }
        }
    }
}

// ---------------- attention pass A: row max + sumexp (online) ----------------
__global__ __launch_bounds__(32) void attn_stats(const unsigned short* __restrict__ Q,
                                                 const unsigned short* __restrict__ Kmat,
                                                 float* __restrict__ Mbuf,
                                                 float* __restrict__ Lbuf) {
    const int b  = blockIdx.y;
    const int qt = blockIdx.x;
    const int q0 = qt * 16;
    const int lane = threadIdx.x & 31;
    const int nn = lane & 15;
    const int mo = (lane & 16) ? 8 : 0;

    const unsigned short* Qb  = Q    + ((size_t)b * S_LEN + q0) * DK;
    const unsigned short* Kb0 = Kmat + (size_t)b * S_LEN * DK;

    float mrun[8], lrun[8];
    for (int i = 0; i < 8; ++i) { mrun[i] = -__builtin_inff(); lrun[i] = 0.0f; }

    int jt = 0;
    // 4-wide key-tile chunks, strictly below the diagonal tile (no masking needed)
    for (; jt + 4 <= qt; jt += 4) {
        const unsigned short* Kb = Kb0 + (size_t)(jt * 16) * DK;
        f32x8 c0 = {}, c1 = {}, c2 = {}, c3 = {};
        for (int kc = 0; kc < DK; kc += 32) {
            bf16x16 a = load_frag(Qb + kc, DK);
            c0 = wmma_bf16(a, load_frag(Kb +  0 * DK + kc, DK), c0);
            c1 = wmma_bf16(a, load_frag(Kb + 16 * DK + kc, DK), c1);
            c2 = wmma_bf16(a, load_frag(Kb + 32 * DK + kc, DK), c2);
            c3 = wmma_bf16(a, load_frag(Kb + 48 * DK + kc, DK), c3);
        }
        for (int i = 0; i < 8; ++i) {
            const float s0 = c0[i] * 0.03125f, s1 = c1[i] * 0.03125f;
            const float s2 = c2[i] * 0.03125f, s3 = c3[i] * 0.03125f;
            float t = fmaxf(fmaxf(s0, s1), fmaxf(s2, s3));
            for (int d = 1; d < 16; d <<= 1) t = fmaxf(t, __shfl_xor(t, d, 16));
            const float mnew = fmaxf(mrun[i], t);
            float e = __expf(s0 - mnew) + __expf(s1 - mnew) +
                      __expf(s2 - mnew) + __expf(s3 - mnew);
            for (int d = 1; d < 16; d <<= 1) e += __shfl_xor(e, d, 16);
            lrun[i] = lrun[i] * __expf(mrun[i] - mnew) + e;
            mrun[i] = mnew;
        }
    }
    // remainder tiles (1..4), incl. the masked diagonal tile
    for (; jt <= qt; ++jt) {
        const int j0 = jt * 16;
        const unsigned short* Kb = Kb0 + (size_t)j0 * DK;
        f32x8 c = {};
        for (int kc = 0; kc < DK; kc += 32)
            c = wmma_bf16(load_frag(Qb + kc, DK), load_frag(Kb + kc, DK), c);
        for (int i = 0; i < 8; ++i) {
            float s = c[i] * 0.03125f;                     // 1/sqrt(1024)
            if (jt == qt && (j0 + nn > q0 + mo + i)) s = -__builtin_inff();
            float t = s;
            for (int d = 1; d < 16; d <<= 1) t = fmaxf(t, __shfl_xor(t, d, 16));
            const float mnew = fmaxf(mrun[i], t);
            float e = __expf(s - mnew);
            for (int d = 1; d < 16; d <<= 1) e += __shfl_xor(e, d, 16);
            lrun[i] = lrun[i] * __expf(mrun[i] - mnew) + e;
            mrun[i] = mnew;
        }
    }
    if (nn == 0) {   // lane 0 writes rows 0..7, lane 16 rows 8..15
        for (int i = 0; i < 8; ++i) {
            const size_t idx = (size_t)b * S_LEN + q0 + mo + i;
            Mbuf[idx] = mrun[i];
            Lbuf[idx] = lrun[i];
        }
    }
}

// ---------------- attention pass B: write normalized P (bf16) ----------------
__global__ __launch_bounds__(32) void attn_probs(const unsigned short* __restrict__ Q,
                                                 const unsigned short* __restrict__ Kmat,
                                                 const float* __restrict__ Mbuf,
                                                 const float* __restrict__ Lbuf,
                                                 unsigned short* __restrict__ P) {
    const int b  = blockIdx.y;
    const int qt = blockIdx.x;
    const int q0 = qt * 16;
    const int lane = threadIdx.x & 31;
    const int nn = lane & 15;
    const int mo = (lane & 16) ? 8 : 0;

    const unsigned short* Qb  = Q    + ((size_t)b * S_LEN + q0) * DK;
    const unsigned short* Kb0 = Kmat + (size_t)b * S_LEN * DK;

    float mrow[8], linv[8];
    for (int i = 0; i < 8; ++i) {
        const size_t idx = (size_t)b * S_LEN + q0 + mo + i;
        mrow[i] = Mbuf[idx];
        linv[i] = 1.0f / Lbuf[idx];
    }
    unsigned short* Prow = P + ((size_t)b * S_LEN + q0) * S_LEN;

    int jt = 0;
    for (; jt + 4 <= qt; jt += 4) {            // full chunks strictly below diagonal
        const int j0 = jt * 16;
        const unsigned short* Kb = Kb0 + (size_t)j0 * DK;
        f32x8 c0 = {}, c1 = {}, c2 = {}, c3 = {};
        for (int kc = 0; kc < DK; kc += 32) {
            bf16x16 a = load_frag(Qb + kc, DK);
            c0 = wmma_bf16(a, load_frag(Kb +  0 * DK + kc, DK), c0);
            c1 = wmma_bf16(a, load_frag(Kb + 16 * DK + kc, DK), c1);
            c2 = wmma_bf16(a, load_frag(Kb + 32 * DK + kc, DK), c2);
            c3 = wmma_bf16(a, load_frag(Kb + 48 * DK + kc, DK), c3);
        }
        f32x8 cc[4] = {c0, c1, c2, c3};
        for (int t = 0; t < 4; ++t)
            for (int i = 0; i < 8; ++i) {
                const float p = __expf(cc[t][i] * 0.03125f - mrow[i]) * linv[i];
                Prow[(size_t)(mo + i) * S_LEN + j0 + t * 16 + nn] = f2bf(p);
            }
    }
    for (; jt <= qt; ++jt) {                   // remainder + masked diagonal
        const int j0 = jt * 16;
        const unsigned short* Kb = Kb0 + (size_t)j0 * DK;
        f32x8 c = {};
        for (int kc = 0; kc < DK; kc += 32)
            c = wmma_bf16(load_frag(Qb + kc, DK), load_frag(Kb + kc, DK), c);
        for (int i = 0; i < 8; ++i) {
            float p = __expf(c[i] * 0.03125f - mrow[i]) * linv[i];
            if (jt == qt && (j0 + nn > q0 + mo + i)) p = 0.0f;
            Prow[(size_t)(mo + i) * S_LEN + j0 + nn] = f2bf(p);
        }
    }
    if ((qt & 1) == 0) {  // pad one zero key-tile: makes 32-row PV grouping legal
        const int j0 = (qt + 1) * 16;
        for (int i = 0; i < 8; ++i)
            Prow[(size_t)(mo + i) * S_LEN + j0 + nn] = 0;
    }
}

// ---------------- ctx = P * V  (V stored transposed), 32x64 per wave ---------
__global__ __launch_bounds__(32) void attn_pv(const unsigned short* __restrict__ P,
                                              const unsigned short* __restrict__ Vt,
                                              unsigned short* __restrict__ ctx) {
    const int g  = blockIdx.x;           // 32-row query group
    const int q0 = g * 32;
    const int d0 = blockIdx.y * 64;
    const int b  = blockIdx.z;
    const int lane = threadIdx.x & 31;
    const int nn = lane & 15;
    const int mo = (lane & 16) ? 8 : 0;

    const unsigned short* A0 = P + ((size_t)b * S_LEN + q0) * S_LEN;
    const unsigned short* A1 = A0 + (size_t)16 * S_LEN;
    const unsigned short* Vb = Vt + (size_t)b * DK * S_LEN;
    const int kmax = 32 * (g + 1);       // valid+zero-padded P columns for both subtiles

    f32x8 cT0 = {}, cT1 = {}, cT2 = {}, cT3 = {};
    f32x8 cB0 = {}, cB1 = {}, cB2 = {}, cB3 = {};
    for (int kc = 0; kc < kmax; kc += 32) {
        bf16x16 a0 = load_frag(A0 + kc, S_LEN);
        bf16x16 a1 = load_frag(A1 + kc, S_LEN);
        bf16x16 b0 = load_frag(Vb + (size_t)(d0 +  0) * S_LEN + kc, S_LEN);
        bf16x16 b1 = load_frag(Vb + (size_t)(d0 + 16) * S_LEN + kc, S_LEN);
        bf16x16 b2 = load_frag(Vb + (size_t)(d0 + 32) * S_LEN + kc, S_LEN);
        bf16x16 b3 = load_frag(Vb + (size_t)(d0 + 48) * S_LEN + kc, S_LEN);
        cT0 = wmma_bf16(a0, b0, cT0);  cT1 = wmma_bf16(a0, b1, cT1);
        cT2 = wmma_bf16(a0, b2, cT2);  cT3 = wmma_bf16(a0, b3, cT3);
        cB0 = wmma_bf16(a1, b0, cB0);  cB1 = wmma_bf16(a1, b1, cB1);
        cB2 = wmma_bf16(a1, b2, cB2);  cB3 = wmma_bf16(a1, b3, cB3);
    }
    f32x8 ccT[4] = {cT0, cT1, cT2, cT3};
    f32x8 ccB[4] = {cB0, cB1, cB2, cB3};
    for (int t = 0; t < 4; ++t)
        for (int i = 0; i < 8; ++i) {
            const int dcol = d0 + t * 16 + nn;
            ctx[((size_t)b * S_LEN + q0 +      mo + i) * DK + dcol] = f2bf(ccT[t][i]);
            ctx[((size_t)b * S_LEN + q0 + 16 + mo + i) * DK + dcol] = f2bf(ccB[t][i]);
        }
}

// ---------------- host-side orchestration ----------------
extern "C" void kernel_launch(void* const* d_in, const int* in_sizes, int n_in,
                              void* d_out, int out_size, void* d_ws, size_t ws_size,
                              hipStream_t stream) {
    (void)in_sizes; (void)n_in; (void)out_size; (void)ws_size;
    const float* x  = (const float*)d_in[0];
    // d_in[1] = mask (tril) -- causality implemented directly
    const float* Wq = (const float*)d_in[2];
    const float* bq = (const float*)d_in[3];
    const float* Wk = (const float*)d_in[4];
    const float* bk = (const float*)d_in[5];
    const float* Wv = (const float*)d_in[6];
    const float* bv = (const float*)d_in[7];
    const float* Wo = (const float*)d_in[8];
    const float* bo = (const float*)d_in[9];

    char* ws = (char*)d_ws;
    size_t off = 0;
    auto alloc = [&](size_t bytes) { char* p = ws + off; off = (off + bytes + 255) & ~(size_t)255; return p; };

    unsigned short* xb   = (unsigned short*)alloc((size_t)M_TOT * DK * 2);
    unsigned short* wqb  = (unsigned short*)alloc((size_t)DK * DK * 2);
    unsigned short* wkb  = (unsigned short*)alloc((size_t)DK * DK * 2);
    unsigned short* wvb  = (unsigned short*)alloc((size_t)DK * DK * 2);
    unsigned short* wob  = (unsigned short*)alloc((size_t)DK * DK * 2);
    unsigned short* Qb   = (unsigned short*)alloc((size_t)M_TOT * DK * 2);
    unsigned short* Kb   = (unsigned short*)alloc((size_t)M_TOT * DK * 2);
    unsigned short* Vt   = (unsigned short*)alloc((size_t)M_TOT * DK * 2);
    float*          Mbuf = (float*)alloc((size_t)M_TOT * 4);
    float*          Lbuf = (float*)alloc((size_t)M_TOT * 4);
    unsigned short* P    = (unsigned short*)alloc((size_t)B_SZ * S_LEN * S_LEN * 2);
    unsigned short* ctx  = (unsigned short*)alloc((size_t)M_TOT * DK * 2);

    // 1) casts
    cast_f32_bf16<<<8192, 256, 0, stream>>>(x,  xb,  (long long)M_TOT * DK);
    cast_f32_bf16<<<1024, 256, 0, stream>>>(Wq, wqb, (long long)DK * DK);
    cast_f32_bf16<<<1024, 256, 0, stream>>>(Wk, wkb, (long long)DK * DK);
    cast_f32_bf16<<<1024, 256, 0, stream>>>(Wv, wvb, (long long)DK * DK);
    cast_f32_bf16<<<1024, 256, 0, stream>>>(Wo, wob, (long long)DK * DK);

    // 2) projections (128x64 tiles, LDS-staged B, async double-buffer)
    dim3 ggrid(M_TOT / 128, DK / 64);
    gemm128x64<<<ggrid, 256, 0, stream>>>(xb, wqb, bq, Qb, DK, DK, 0);
    gemm128x64<<<ggrid, 256, 0, stream>>>(xb, wkb, bk, Kb, DK, DK, 0);
    gemm128x64<<<ggrid, 256, 0, stream>>>(xb, wvb, bv, Vt, DK, DK, 1);  // transposed V

    // 3) softmax statistics, 4) probabilities
    dim3 agrid(S_LEN / 16, B_SZ);
    attn_stats<<<agrid, 32, 0, stream>>>(Qb, Kb, Mbuf, Lbuf);
    attn_probs<<<agrid, 32, 0, stream>>>(Qb, Kb, Mbuf, Lbuf, P);

    // 5) ctx = P * V
    dim3 pvgrid(S_LEN / 32, DK / 64, B_SZ);
    attn_pv<<<pvgrid, 32, 0, stream>>>(P, Vt, ctx);

    // 6) out = ctx * Wo^T + bo (fp32)
    gemm128x64<<<ggrid, 256, 0, stream>>>(ctx, wob, bo, d_out, DK, DK, 2);
}
